// MoEPolicy_78709570667040
// MI455X (gfx1250) — matrix operand order repaired and checked by reading, not compiled, not run on hardware
//
#include <hip/hip_runtime.h>

#define NROWS  16384
#define DDIM   256
#define HDIM   1024
#define NEXP   16
#define KSH    2
#define NGR    64
#define GH     128
#define MT     32
#define NZ     (KSH + NEXP)

// padded LDS leading dims (rows land 4 banks apart -> conflict-free ds_load_b128)
#define XLD    (DDIM + 8)    // bf16 elems, 528B row stride, 16B aligned
#define HLD    (HDIM + 8)    // bf16 elems, 2064B row stride, 16B aligned
#define YLD    (DDIM + 4)    // f32 elems, 1040B row stride

typedef __attribute__((ext_vector_type(16))) __bf16 v16bf;
typedef __attribute__((ext_vector_type(8)))  float  v8f;

// ---------- bf16 helpers (raw-bit storage so host never sees __bf16) ----------
__device__ __forceinline__ unsigned short f2bfu(float f) {
  union { float f; unsigned u; } v; v.f = f;
  unsigned r = v.u + 0x7FFFu + ((v.u >> 16) & 1u);   // round-to-nearest-even
  return (unsigned short)(r >> 16);
}

// branch-free GELU (tanh form): v_exp_f32 + v_rcp, no exec-mask divergence
__device__ __forceinline__ float gelu_f(float x) {
  float u = 0.7978845608028654f * x * (1.0f + 0.044715f * x * x);
  float e = __expf(2.0f * u);
  float th = 1.0f - 2.0f / (e + 1.0f);               // tanh(u), saturates correctly
  return 0.5f * x * (1.0f + th);
}

// ---------- WMMA fragment loaders (CDNA5 wave32 layouts, ISA 7.12.2) ----------
// A: 16x32 bf16 from padded LDS. Per lane: two contiguous 16B chunks ->
//    elems 0-7 = K base..base+7, elems 8-15 = K base+16..base+23, base = k0 + (lane>=16)*8
__device__ __forceinline__ v16bf load_a_frag(const unsigned short* p) {
  uint4 lo = *(const uint4*)p;
  uint4 hi = *(const uint4*)(p + 16);
  v16bf a;
  __builtin_memcpy(&a, &lo, 16);
  __builtin_memcpy((char*)&a + 16, &hi, 16);
  return a;
}
// B: packed fragment-major (32B contiguous per lane) -> two global_load_b128
__device__ __forceinline__ v16bf load_b_packed(const unsigned short* p) {
  v16bf b;
  __builtin_memcpy(&b, p, 32);
  return b;
}

// ------------------------------- small kernels -------------------------------
__global__ void zero_kernel(float* p, int n) {
  int i = blockIdx.x * blockDim.x + threadIdx.x;
  if (i < n) p[i] = 0.0f;
}

// pack fp32 weight matrices [count][K][N] into bf16 WMMA-B fragment-major layout:
// P[ex][ (nt*ktiles + kt)*32 + lane ][e] = bf( W[ex][ kt*32 + (lane>=16)*8 + e + (e>=8)*8 ][ nt*16 + (lane&15) ] )
__global__ void pack_b_kernel(const float* __restrict__ W, unsigned short* __restrict__ P,
                              int K, int N, int count) {
  int idx = blockIdx.x * blockDim.x + threadIdx.x;
  int ktiles = K / 32, ntiles = N / 16;
  int total = count * ntiles * ktiles * 32;
  if (idx >= total) return;
  int lane = idx & 31;
  int t2 = idx >> 5;
  int kt = t2 % ktiles;
  int t3 = t2 / ktiles;
  int nt = t3 % ntiles;
  int ex = t3 / ntiles;
  const float* Wb = W + (size_t)ex * K * N;
  unsigned short* Pb = P + (size_t)ex * K * N +
                       ((((size_t)nt * ktiles + kt) * 32 + lane) << 4);
  int half = lane >> 4;
  int n = nt * 16 + (lane & 15);
#pragma unroll
  for (int e = 0; e < 16; ++e)
    Pb[e] = f2bfu(Wb[(size_t)(kt * 32 + half * 8 + e + (e >= 8 ? 8 : 0)) * N + n]);
}

// per-node: acc = v_emb (residual init), pooled segment sums, bf16 copy of x
__global__ void pool_init_kernel(const float* __restrict__ x, const int* __restrict__ bidx,
                                 float* __restrict__ gsum, float* __restrict__ counts,
                                 float* __restrict__ acc, unsigned short* __restrict__ xbf) {
  int n = blockIdx.x;
  int d = threadIdx.x;
  int b = bidx[n];
  float v = x[(size_t)n * DDIM + d];
  acc[(size_t)n * DDIM + d] = v;
  xbf[(size_t)n * DDIM + d] = f2bfu(v);
  atomicAdd(&gsum[b * DDIM + d], v);
  if (d == 0) atomicAdd(&counts[b], 1.0f);
}

// gating: one thread per graph (tiny)
__global__ void gating_kernel(const float* __restrict__ gsum, const float* __restrict__ counts,
                              const float* __restrict__ gw1, const float* __restrict__ gb1,
                              const float* __restrict__ gw2, const float* __restrict__ gb2,
                              const float* __restrict__ alpha, const float* __restrict__ ebias,
                              float* __restrict__ route) {
  int b = threadIdx.x;
  if (b >= NGR) return;
  float inv = 1.0f / fmaxf(counts[b], 1.0f);
  float hg[GH];
  for (int j = 0; j < GH; ++j) {
    float s = gb1[j];
    for (int d = 0; d < DDIM; ++d) s += gsum[b * DDIM + d] * inv * gw1[d * GH + j];
    hg[j] = (s >= 0.0f) ? s : 0.2f * s;   // leaky relu
  }
  float A = alpha[0] / 0.6f;              // alpha / TEMP
  float lg[NEXP];
  for (int e = 0; e < NEXP; ++e) {
    float s = gb2[e];
    for (int j = 0; j < GH; ++j) s += hg[j] * gw2[j * NEXP + e];
    lg[e] = s * A + ebias[e];
  }
  bool sel[NEXP];
  for (int e = 0; e < NEXP; ++e) sel[e] = false;
  for (int t = 0; t < 4; ++t) {           // top-4, first-index tie-break like lax.top_k
    int am = 0; float mv = -3.402823466e38f;
    for (int e = 0; e < NEXP; ++e)
      if (!sel[e] && lg[e] > mv) { mv = lg[e]; am = e; }
    sel[am] = true;
  }
  float m = lg[0];
  for (int e = 1; e < NEXP; ++e) m = fmaxf(m, lg[e]);
  float S = 0.0f, w[NEXP];
  for (int e = 0; e < NEXP; ++e) { w[e] = expf(lg[e] - m); S += w[e]; }
  float sw = 0.0f;
  for (int e = 0; e < NEXP; ++e) { w[e] = sel[e] ? w[e] / S : 0.0f; sw += w[e]; }
  float invs = 1.0f / (sw + 1e-12f);
  for (int e = 0; e < NEXP; ++e) route[b * NEXP + e] = w[e] * invs;
}

// ------------------- fused two-layer expert (WMMA bf16) ----------------------
// grid.x: row tile (MT rows), grid.y: z in [0,18): 0..1 shared experts, 2..17 dedicated
// Each wave computes 2x2 macro-tiles (32x32 outputs): 4 WMMAs per 2 A-loads + 2 B-loads.
__global__ __launch_bounds__(256)
void expert_kernel(const unsigned short* __restrict__ Xbf,
                   const unsigned short* __restrict__ sw1p, const unsigned short* __restrict__ sw2p,
                   const unsigned short* __restrict__ dw1p, const unsigned short* __restrict__ dw2p,
                   const float* __restrict__ sb1, const float* __restrict__ sb2,
                   const float* __restrict__ sg,  const float* __restrict__ sbeta,
                   const float* __restrict__ db1, const float* __restrict__ db2,
                   const float* __restrict__ dg,  const float* __restrict__ dbeta,
                   const int* __restrict__ bidx,  const float* __restrict__ route,
                   float* __restrict__ acc) {
  extern __shared__ char smem[];
  unsigned short* Xs = (unsigned short*)smem;                                 // MT*XLD bf16
  unsigned short* Hs = (unsigned short*)(smem + MT * XLD * 2);                // MT*HLD bf16
  float* Ys    = (float*)(smem + MT * XLD * 2 + MT * HLD * 2);                // MT*YLD f32
  float* Stats = (float*)(smem + MT * XLD * 2 + MT * HLD * 2 + MT * YLD * 4); // 2*MT f32

  const int z    = blockIdx.y;
  const int row0 = blockIdx.x * MT;
  const int tid  = threadIdx.x;
  const int lane = tid & 31;
  const int wv   = tid >> 5;
  const int half = lane >> 4;
  const int ln15 = lane & 15;

  const unsigned short *W1, *W2;
  const float *B1, *B2, *G, *Beta;
  if (z < KSH) {
    W1 = sw1p + (size_t)z * DDIM * HDIM;  W2 = sw2p + (size_t)z * HDIM * DDIM;
    B1 = sb1 + z * HDIM;  B2 = sb2 + z * DDIM;  G = sg + z * DDIM;  Beta = sbeta + z * DDIM;
  } else {
    int e = z - KSH;
    W1 = dw1p + (size_t)e * DDIM * HDIM;  W2 = dw2p + (size_t)e * HDIM * DDIM;
    B1 = db1 + e * HDIM;  B2 = db2 + e * DDIM;  G = dg + e * DDIM;  Beta = dbeta + e * DDIM;
  }

  // ---- stage X tile into (padded) LDS via CDNA5 async LDS-DMA: 16B per lane,
  //      no VGPR round trip, tracked by ASYNCcnt.
  {
    const unsigned short* xsrc = Xbf + (size_t)row0 * DDIM;
    for (int ch = tid; ch < (MT * DDIM) / 8; ch += 256) {     // 16B chunks
      int r  = ch >> 5;                // 32 chunks per 256-elem row
      int c8 = (ch & 31) * 8;
      unsigned lds_off = (unsigned)((r * XLD + c8) * 2);
      const unsigned short* gp = xsrc + (size_t)r * DDIM + c8;
      asm volatile("global_load_async_to_lds_b128 %0, %1, off"
                   :: "v"(lds_off), "v"(gp) : "memory");
    }
    asm volatile("s_wait_asynccnt 0x0" ::: "memory");
  }
  __syncthreads();

  // ---- layer 1: Hs = gelu(Xs @ W1 + b1); 32 col-pairs, 4 macro-tiles per wave
  const int KT1 = DDIM / 32;
  for (int cp = wv; cp < HDIM / 32; cp += 8) {
    const int n0 = cp * 32;                  // col tiles n0 and n0+16
    const float bv0 = B1[n0 + ln15];
    const float bv1 = B1[n0 + 16 + ln15];
    v8f c00, c01, c10, c11;
#pragma unroll
    for (int j = 0; j < 8; ++j) { c00[j] = bv0; c10[j] = bv0; c01[j] = bv1; c11[j] = bv1; }
    const unsigned short* pa0 = Xs + (size_t)ln15 * XLD + half * 8;          // rows 0..15
    const unsigned short* pa1 = Xs + (size_t)(16 + ln15) * XLD + half * 8;   // rows 16..31
    const unsigned short* pb0 = W1 + ((((size_t)(2 * cp)     * KT1) * 32 + lane) << 4);
    const unsigned short* pb1 = W1 + ((((size_t)(2 * cp + 1) * KT1) * 32 + lane) << 4);
    for (int kt = 0; kt < KT1; ++kt) {
      __builtin_prefetch(pb0 + 1024, 0, 1);
      __builtin_prefetch(pb1 + 1024, 0, 1);
      v16bf a0 = load_a_frag(pa0);
      v16bf a1 = load_a_frag(pa1);
      v16bf b0 = load_b_packed(pb0);
      v16bf b1 = load_b_packed(pb1);
      c00 = __builtin_amdgcn_wmma_f32_16x16x32_bf16(false, a0, false, b0, (short)0, c00, false, false);
      c01 = __builtin_amdgcn_wmma_f32_16x16x32_bf16(false, a0, false, b1, (short)0, c01, false, false);
      c10 = __builtin_amdgcn_wmma_f32_16x16x32_bf16(false, a1, false, b0, (short)0, c10, false, false);
      c11 = __builtin_amdgcn_wmma_f32_16x16x32_bf16(false, a1, false, b1, (short)0, c11, false, false);
      pa0 += 32; pa1 += 32; pb0 += 512; pb1 += 512;
    }
    const int rb0 = half * 8;          // rows 0..15 half
    const int rb1 = 16 + half * 8;     // rows 16..31 half
#pragma unroll
    for (int j = 0; j < 8; ++j) {
      Hs[(rb0 + j) * HLD + n0 + ln15]      = f2bfu(gelu_f(c00[j]));
      Hs[(rb0 + j) * HLD + n0 + 16 + ln15] = f2bfu(gelu_f(c01[j]));
      Hs[(rb1 + j) * HLD + n0 + ln15]      = f2bfu(gelu_f(c10[j]));
      Hs[(rb1 + j) * HLD + n0 + 16 + ln15] = f2bfu(gelu_f(c11[j]));
    }
  }
  __syncthreads();

  // ---- layer 2: Ys = Hs @ W2 + b2; 8 col-pairs, 1 macro-tile per wave
  const int KT2 = HDIM / 32;
  for (int cp = wv; cp < DDIM / 32; cp += 8) {
    const int n0 = cp * 32;
    const float bv0 = B2[n0 + ln15];
    const float bv1 = B2[n0 + 16 + ln15];
    v8f c00, c01, c10, c11;
#pragma unroll
    for (int j = 0; j < 8; ++j) { c00[j] = bv0; c10[j] = bv0; c01[j] = bv1; c11[j] = bv1; }
    const unsigned short* pa0 = Hs + (size_t)ln15 * HLD + half * 8;
    const unsigned short* pa1 = Hs + (size_t)(16 + ln15) * HLD + half * 8;
    const unsigned short* pb0 = W2 + ((((size_t)(2 * cp)     * KT2) * 32 + lane) << 4);
    const unsigned short* pb1 = W2 + ((((size_t)(2 * cp + 1) * KT2) * 32 + lane) << 4);
    for (int kt = 0; kt < KT2; ++kt) {
      __builtin_prefetch(pb0 + 1024, 0, 1);
      __builtin_prefetch(pb1 + 1024, 0, 1);
      v16bf a0 = load_a_frag(pa0);
      v16bf a1 = load_a_frag(pa1);
      v16bf b0 = load_b_packed(pb0);
      v16bf b1 = load_b_packed(pb1);
      c00 = __builtin_amdgcn_wmma_f32_16x16x32_bf16(false, a0, false, b0, (short)0, c00, false, false);
      c01 = __builtin_amdgcn_wmma_f32_16x16x32_bf16(false, a0, false, b1, (short)0, c01, false, false);
      c10 = __builtin_amdgcn_wmma_f32_16x16x32_bf16(false, a1, false, b0, (short)0, c10, false, false);
      c11 = __builtin_amdgcn_wmma_f32_16x16x32_bf16(false, a1, false, b1, (short)0, c11, false, false);
      pa0 += 32; pa1 += 32; pb0 += 512; pb1 += 512;
    }
    const int rb0 = half * 8;
    const int rb1 = 16 + half * 8;
#pragma unroll
    for (int j = 0; j < 8; ++j) {
      Ys[(rb0 + j) * YLD + n0 + ln15]      = c00[j];
      Ys[(rb0 + j) * YLD + n0 + 16 + ln15] = c01[j];
      Ys[(rb1 + j) * YLD + n0 + ln15]      = c10[j];
      Ys[(rb1 + j) * YLD + n0 + 16 + ln15] = c11[j];
    }
  }
  __syncthreads();

  // ---- LayerNorm stats per row (biased var, eps=1e-5)
  if (tid < MT) {
    const float* yr = Ys + tid * YLD;
    float s = 0.0f, s2 = 0.0f;
    for (int cdx = 0; cdx < DDIM; ++cdx) { float v = yr[cdx]; s += v; s2 += v * v; }
    float mu  = s * (1.0f / DDIM);
    float var = s2 * (1.0f / DDIM) - mu * mu;
    Stats[tid]      = mu;
    Stats[MT + tid] = rsqrtf(var + 1e-5f);
  }
  __syncthreads();

  // ---- weighted accumulate into combined
  {
    const int r  = tid >> 3;            // 0..31
    const int c0 = (tid & 7) * 32;      // 8 threads cover 256 cols
    const int n  = row0 + r;
    float w = (z < KSH) ? (1.0f / KSH) : route[bidx[n] * NEXP + (z - KSH)];
    if (w != 0.0f) {
      const float mu = Stats[r], rs = Stats[MT + r];
      for (int cc = 0; cc < 32; ++cc) {
        int col = c0 + cc;
        float v = (Ys[r * YLD + col] - mu) * rs * G[col] + Beta[col];
        atomicAdd(&acc[(size_t)n * DDIM + col], w * v);
      }
    }
  }
}

// ------------------------------- head --------------------------------------
__global__ void head_kernel(const float* __restrict__ acc, const float* __restrict__ hw,
                            const float* __restrict__ hb, float* __restrict__ out) {
  int n = blockIdx.x * blockDim.x + threadIdx.x;
  if (n >= NROWS) return;
  const float4* a = (const float4*)(acc + (size_t)n * DDIM);
  const float4* w = (const float4*)hw;
  float s = 0.0f;
  for (int i = 0; i < DDIM / 4; ++i) {
    float4 av = a[i], wv = w[i];
    s += av.x * wv.x + av.y * wv.y + av.z * wv.z + av.w * wv.w;
  }
  out[n] = s + hb[0];
}

// ----------------------------- launcher ------------------------------------
extern "C" void kernel_launch(void* const* d_in, const int* in_sizes, int n_in,
                              void* d_out, int out_size, void* d_ws, size_t ws_size,
                              hipStream_t stream) {
  (void)in_sizes; (void)n_in; (void)out_size; (void)ws_size;
  const float* v_emb  = (const float*)d_in[0];
  const int*   bidx   = (const int*)d_in[1];
  const float* gw1    = (const float*)d_in[2];
  const float* gb1    = (const float*)d_in[3];
  const float* gw2    = (const float*)d_in[4];
  const float* gb2    = (const float*)d_in[5];
  const float* alpha  = (const float*)d_in[6];
  const float* ebias  = (const float*)d_in[7];
  const float* sw1    = (const float*)d_in[8];
  const float* sb1    = (const float*)d_in[9];
  const float* sw2    = (const float*)d_in[10];
  const float* sb2    = (const float*)d_in[11];
  const float* sg     = (const float*)d_in[12];
  const float* sbeta  = (const float*)d_in[13];
  const float* dw1    = (const float*)d_in[14];
  const float* db1    = (const float*)d_in[15];
  const float* dw2    = (const float*)d_in[16];
  const float* db2    = (const float*)d_in[17];
  const float* dg     = (const float*)d_in[18];
  const float* dbeta  = (const float*)d_in[19];
  const float* head_w = (const float*)d_in[20];
  const float* head_b = (const float*)d_in[21];
  float* out = (float*)d_out;

  char* ws = (char*)d_ws;
  size_t off = 0;
  auto take = [&](size_t bytes) -> char* {
    char* p = ws + off;
    off = (off + bytes + 255) & ~(size_t)255;
    return p;
  };
  float* counts        = (float*)take(NGR * 4);
  float* gsum          = (float*)take((size_t)NGR * DDIM * 4);
  float* route         = (float*)take((size_t)NGR * NEXP * 4);
  float* acc           = (float*)take((size_t)NROWS * DDIM * 4);
  unsigned short* xbf  = (unsigned short*)take((size_t)NROWS * DDIM * 2);
  unsigned short* sw1p = (unsigned short*)take((size_t)KSH * DDIM * HDIM * 2);
  unsigned short* sw2p = (unsigned short*)take((size_t)KSH * HDIM * DDIM * 2);
  unsigned short* dw1p = (unsigned short*)take((size_t)NEXP * DDIM * HDIM * 2);
  unsigned short* dw2p = (unsigned short*)take((size_t)NEXP * HDIM * DDIM * 2);

  // zero pooled accumulators (re-done every call -> deterministic replays)
  zero_kernel<<<(NGR + 255) / 256, 256, 0, stream>>>(counts, NGR);
  zero_kernel<<<(NGR * DDIM + 255) / 256, 256, 0, stream>>>(gsum, NGR * DDIM);

  // one-time fp32 -> bf16 conversion + WMMA fragment-major repack of weights
  {
    int t1 = KSH  * (HDIM / 16) * (DDIM / 32) * 32;
    int t2 = KSH  * (DDIM / 16) * (HDIM / 32) * 32;
    int t3 = NEXP * (HDIM / 16) * (DDIM / 32) * 32;
    int t4 = NEXP * (DDIM / 16) * (HDIM / 32) * 32;
    pack_b_kernel<<<(t1 + 255) / 256, 256, 0, stream>>>(sw1, sw1p, DDIM, HDIM, KSH);
    pack_b_kernel<<<(t2 + 255) / 256, 256, 0, stream>>>(sw2, sw2p, HDIM, DDIM, KSH);
    pack_b_kernel<<<(t3 + 255) / 256, 256, 0, stream>>>(dw1, dw1p, DDIM, HDIM, NEXP);
    pack_b_kernel<<<(t4 + 255) / 256, 256, 0, stream>>>(dw2, dw2p, HDIM, DDIM, NEXP);
  }

  // pooling + residual init + x downconvert
  pool_init_kernel<<<NROWS, DDIM, 0, stream>>>(v_emb, bidx, gsum, counts, acc, xbf);

  // gating / routing
  gating_kernel<<<1, 64, 0, stream>>>(gsum, counts, gw1, gb1, gw2, gb2, alpha, ebias, route);

  // fused experts (2 shared + 16 dedicated) accumulating into acc
  size_t shbytes = (size_t)MT * XLD * 2 + (size_t)MT * HLD * 2 +
                   (size_t)MT * YLD * 4 + 2 * MT * 4;   // ~116 KB of the 320 KB WGP LDS
  expert_kernel<<<dim3(NROWS / MT, NZ), 256, shbytes, stream>>>(
      xbf, sw1p, sw2p, dw1p, dw2p, sb1, sb2, sg, sbeta,
      db1, db2, dg, dbeta, bidx, route, acc);

  // readout head
  head_kernel<<<(NROWS + 255) / 256, 256, 0, stream>>>(acc, head_w, head_b, out);
}
